// VRGNNTransformer_80221399155599
// MI455X (gfx1250) — matrix operand order repaired and checked by reading, not compiled, not run on hardware
//
#include <hip/hip_runtime.h>

// ---------------------------------------------------------------------------
// CDNA5 (gfx1250, wave32) hetero-SAGE GNN + Transformer.
// GEMM: v_wmma_f32_16x16x32_bf16, fragment-ordered LDS staging (per-lane
// operand = contiguous 32B -> 2x ds_load_b128), double-buffered pipeline with
// an LDS-only barrier; FF intermediate kept in bf16 to halve HBM traffic.
// ---------------------------------------------------------------------------

typedef __attribute__((ext_vector_type(16))) __bf16 v16bf;
typedef __attribute__((ext_vector_type(4)))  __bf16 v4bf;
typedef __attribute__((ext_vector_type(8)))  float  v8f;

#define TB 256

// LDS-only barrier: wait for our DS stores, then split workgroup barrier.
// Crucially does NOT wait on outstanding global (LOADcnt) prefetches.
__device__ __forceinline__ void lds_barrier() {
  asm volatile(
      "s_wait_dscnt 0x0\n\t"
      "s_barrier_signal -1\n\t"
      "s_barrier_wait -1" ::: "memory");
}

// ---------------- zero ----------------
__global__ void k_zero(float* __restrict__ p, long long n) {
  long long i = (long long)blockIdx.x * TB + threadIdx.x;
  if (i < n) p[i] = 0.0f;
}

// ---------------- SAGE edge aggregation (gather + atomic scatter-add) ------
// din is a power of two (128 or 64): shift = log2(din)
__global__ void k_sage_agg(const float* __restrict__ xsrc, const int* __restrict__ ei,
                           float* __restrict__ agg, float* __restrict__ cnt,
                           int E, int shift) {
  long long idx = (long long)blockIdx.x * TB + threadIdx.x;
  long long total = (long long)E << shift;
  if (idx >= total) return;
  int e = (int)(idx >> shift);
  int f = (int)(idx & ((1 << shift) - 1));
  int s = ei[e];          // src row
  int d = ei[E + e];      // dst row
  atomicAdd(&agg[((unsigned)d << shift) + f], xsrc[((unsigned)s << shift) + f]);
  if (f == 0) atomicAdd(&cnt[d], 1.0f);
}

__global__ void k_mean(float* __restrict__ agg, const float* __restrict__ cnt,
                       int shift, long long total) {
  long long idx = (long long)blockIdx.x * TB + threadIdx.x;
  if (idx >= total) return;
  int node = (int)(idx >> shift);
  agg[idx] = agg[idx] / fmaxf(cnt[node], 1.0f);
}

// ---------------- bf16 WMMA GEMM -------------------------------------------
// C[M,N] (+)= A[M,K] * B (+ bias) (relu?)
//   BT==0 : B is [K,N] row-major ; BT==1 : B is [N,K] row-major (A @ B^T)
//   ABF==1: A is bf16 in memory (no conversion in the hot loop)
//   CBF==1: C stored as bf16
// Block tile 64 x (64*NB), BK=32, 8 waves; each wave owns 2*NB 16x16 C tiles
// -> 2*NB WMMAs per wave per k-step. Double-buffered LDS pipeline, one
// LDS-only barrier per k-step.
// Requires: M % 64 == 0, N % (64*NB) == 0, K % 32 == 0, K >= 64.
#define BM 64
#define BK 32

template <int BT, int NB, int ABF, int CBF>
__global__ __launch_bounds__(256)
void k_gemm_bf16(const void* __restrict__ Av,
                 const float* __restrict__ B, int ldb,
                 const float* __restrict__ bias,
                 void* __restrict__ Cv,
                 int N, int K, int accumulate, int relu) {
  constexpr int BNt = 64 * NB;
  const float*  Af = (const float*)Av;
  const __bf16* Ah = (const __bf16*)Av;
  float*  Cf = (float*)Cv;
  __bf16* Ch = (__bf16*)Cv;

  // Fragment-ordered staging: frag[buf][blk*512 + lane*16 + e] (bf16):
  // each lane's 16-element WMMA operand is one contiguous 32B chunk.
  __shared__ alignas(32) __bf16 fragA[2][2048];        // 4 m-blocks of 16 rows
  __shared__ alignas(32) __bf16 fragB[2][2048 * NB];   // 4*NB n-blocks of 16 cols

  const int tid  = threadIdx.x;
  const int lane = tid & 31;
  const int wave = tid >> 5;
  const int wr   = wave >> 2;      // 0..1 -> m-blocks 2*wr, 2*wr+1
  const int wc   = wave & 3;       // n-blocks wc, wc+4, ...
  const int m0   = blockIdx.x * BM;
  const int n0   = blockIdx.y * BNt;

  // --- per-thread loader geometry (k0-invariant, 32-bit element offsets) ---
  // A tile: 64 rows x 32 k = 512 quads of 4; 2 per thread.
  unsigned offA[2]; int ldsA[2];
#pragma unroll
  for (int i = 0; i < 2; ++i) {
    int q  = i * 256 + tid;
    int r  = q >> 3;               // row 0..63
    int kq = (q & 7) * 4;          // k quad base 0..28
    int half = kq >> 4;
    int e  = kq & 15;
    ldsA[i] = (r >> 4) * 512 + (half * 16 + (r & 15)) * 16 + e;
    offA[i] = (unsigned)(m0 + r) * (unsigned)K + (unsigned)kq;
  }
  // B tile: 32 k x (64*NB) n = 512*NB quads; 2*NB per thread.
  unsigned offB[2 * NB]; int ldsB[2 * NB];
#pragma unroll
  for (int i = 0; i < 2 * NB; ++i) {
    int q = i * 256 + tid;
    if (BT) {                      // B[n][k]: quad along k (contiguous)
      int n  = q >> 3;             // 0..BNt-1
      int kq = (q & 7) * 4;
      ldsB[i] = (n >> 4) * 512 + ((kq >> 4) * 16 + (n & 15)) * 16 + (kq & 15);
      offB[i] = (unsigned)(n0 + n) * (unsigned)ldb + (unsigned)kq;
    } else {                       // B[k][n]: quad along n (contiguous)
      int k  = q / (16 * NB);      // 0..31
      int nq = (q % (16 * NB)) * 4;
      ldsB[i] = (nq >> 4) * 512 + ((k >> 4) * 16 + (nq & 15)) * 16 + (k & 15);
      offB[i] = (unsigned)k * (unsigned)ldb + (unsigned)(n0 + nq);
    }
  }
  const unsigned stepB = BT ? (unsigned)BK : (unsigned)BK * (unsigned)ldb;

  const int fa0 = (2 * wr + 0) * 512 + lane * 16;
  const int fa1 = (2 * wr + 1) * 512 + lane * 16;
  int fb[NB];
#pragma unroll
  for (int j = 0; j < NB; ++j) fb[j] = (wc + 4 * j) * 512 + lane * 16;

  // ---- prologue: stage tile 0 into buffer 0 ----
  {
    if (ABF) {
      *(v4bf*)&fragA[0][ldsA[0]] = *(const v4bf*)(Ah + offA[0]);
      *(v4bf*)&fragA[0][ldsA[1]] = *(const v4bf*)(Ah + offA[1]);
    } else {
      float4 a0 = *(const float4*)(Af + offA[0]);
      float4 a1 = *(const float4*)(Af + offA[1]);
      *(v4bf*)&fragA[0][ldsA[0]] = (v4bf){ (__bf16)a0.x, (__bf16)a0.y, (__bf16)a0.z, (__bf16)a0.w };
      *(v4bf*)&fragA[0][ldsA[1]] = (v4bf){ (__bf16)a1.x, (__bf16)a1.y, (__bf16)a1.z, (__bf16)a1.w };
    }
    float4 bv[2 * NB];
#pragma unroll
    for (int i = 0; i < 2 * NB; ++i) bv[i] = *(const float4*)(B + offB[i]);
#pragma unroll
    for (int i = 0; i < 2 * NB; ++i) {
      if (BT) {
        *(v4bf*)&fragB[0][ldsB[i]] =
            (v4bf){ (__bf16)bv[i].x, (__bf16)bv[i].y, (__bf16)bv[i].z, (__bf16)bv[i].w };
      } else {
        __bf16* d = &fragB[0][ldsB[i]];
        d[0] = (__bf16)bv[i].x; d[16] = (__bf16)bv[i].y;
        d[32] = (__bf16)bv[i].z; d[48] = (__bf16)bv[i].w;
      }
    }
  }

  v8f acc[2][NB];
#pragma unroll
  for (int t = 0; t < 2; ++t)
#pragma unroll
    for (int j = 0; j < NB; ++j) acc[t][j] = (v8f){};

  int buf = 0;
  for (int k0 = BK; k0 < K; k0 += BK) {
    // next-tile global prefetch (register destinations; not flushed at barrier)
    float4 a0f{}, a1f{};
    v4bf   a0h{}, a1h{};
    if (ABF) {
      a0h = *(const v4bf*)(Ah + offA[0] + (unsigned)k0);
      a1h = *(const v4bf*)(Ah + offA[1] + (unsigned)k0);
    } else {
      a0f = *(const float4*)(Af + offA[0] + (unsigned)k0);
      a1f = *(const float4*)(Af + offA[1] + (unsigned)k0);
    }
    unsigned bo = (unsigned)(k0 >> 5) * stepB;
    float4 bv[2 * NB];
#pragma unroll
    for (int i = 0; i < 2 * NB; ++i) bv[i] = *(const float4*)(B + offB[i] + bo);

    lds_barrier();                 // stage of current tile visible (DS only)

    v16bf fA0 = *(const v16bf*)&fragA[buf][fa0];
    v16bf fA1 = *(const v16bf*)&fragA[buf][fa1];
#pragma unroll
    for (int j = 0; j < NB; ++j) {
      v16bf fB = *(const v16bf*)&fragB[buf][fb[j]];
      acc[0][j] = __builtin_amdgcn_wmma_f32_16x16x32_bf16(
          false, fA0, false, fB, (short)0, acc[0][j], false, false);
      acc[1][j] = __builtin_amdgcn_wmma_f32_16x16x32_bf16(
          false, fA1, false, fB, (short)0, acc[1][j], false, false);
    }

    const int nb = buf ^ 1;        // stage next tile into the other buffer
    if (ABF) {
      *(v4bf*)&fragA[nb][ldsA[0]] = a0h;
      *(v4bf*)&fragA[nb][ldsA[1]] = a1h;
    } else {
      *(v4bf*)&fragA[nb][ldsA[0]] = (v4bf){ (__bf16)a0f.x, (__bf16)a0f.y, (__bf16)a0f.z, (__bf16)a0f.w };
      *(v4bf*)&fragA[nb][ldsA[1]] = (v4bf){ (__bf16)a1f.x, (__bf16)a1f.y, (__bf16)a1f.z, (__bf16)a1f.w };
    }
#pragma unroll
    for (int i = 0; i < 2 * NB; ++i) {
      if (BT) {
        *(v4bf*)&fragB[nb][ldsB[i]] =
            (v4bf){ (__bf16)bv[i].x, (__bf16)bv[i].y, (__bf16)bv[i].z, (__bf16)bv[i].w };
      } else {
        __bf16* d = &fragB[nb][ldsB[i]];
        d[0] = (__bf16)bv[i].x; d[16] = (__bf16)bv[i].y;
        d[32] = (__bf16)bv[i].z; d[48] = (__bf16)bv[i].w;
      }
    }
    buf = nb;
  }

  lds_barrier();                   // final tile
  {
    v16bf fA0 = *(const v16bf*)&fragA[buf][fa0];
    v16bf fA1 = *(const v16bf*)&fragA[buf][fa1];
#pragma unroll
    for (int j = 0; j < NB; ++j) {
      v16bf fB = *(const v16bf*)&fragB[buf][fb[j]];
      acc[0][j] = __builtin_amdgcn_wmma_f32_16x16x32_bf16(
          false, fA0, false, fB, (short)0, acc[0][j], false, false);
      acc[1][j] = __builtin_amdgcn_wmma_f32_16x16x32_bf16(
          false, fA1, false, fB, (short)0, acc[1][j], false, false);
    }
  }

  // epilogue: C layout = VGPR v -> row (half*8 + v) of the 16x16 tile
  const int half = lane >> 4;
#pragma unroll
  for (int j = 0; j < NB; ++j) {
    const int nn = n0 + (wc + 4 * j) * 16 + (lane & 15);
    const float bval = bias ? bias[nn] : 0.0f;
#pragma unroll
    for (int t = 0; t < 2; ++t) {
      unsigned base = (unsigned)(m0 + (2 * wr + t) * 16 + half * 8) * (unsigned)N + (unsigned)nn;
#pragma unroll
      for (int v = 0; v < 8; ++v) {
        unsigned off = base + (unsigned)v * (unsigned)N;
        float val = acc[t][j][v] + bval;
        if (!CBF && accumulate) val += Cf[off];
        if (relu) val = fmaxf(val, 0.0f);
        if (CBF) Ch[off] = (__bf16)val;
        else     Cf[off] = val;
      }
    }
  }
}

// ---------------- concat(xL,xR) + [b,s]->[s,b] permute ---------------------
__global__ void k_concat_permute(const float* __restrict__ xl,
                                 const float* __restrict__ xr,
                                 float* __restrict__ hT, int Nn, int seq) {
  long long idx = (long long)blockIdx.x * TB + threadIdx.x;
  if (idx >= (long long)Nn * 128) return;
  int n = (int)(idx >> 7);
  int c = (int)(idx & 127);
  float v = (c < 64) ? xl[(unsigned)n * 64 + c] : xr[(unsigned)n * 64 + (c - 64)];
  int b = n / seq;
  int s = n - b * seq;
  hT[(unsigned)(s * 32 + b) * 128u + (unsigned)c] = v;
}

// ---------------- fused MHA (groups of 32 tokens, 8 heads, DH=16) ----------
__global__ __launch_bounds__(256)
void k_attn(const float* __restrict__ qkv, float* __restrict__ out) {
  __shared__ float Ks[32][129];
  __shared__ float Vs[32][129];
  const int g = blockIdx.x;      // attention group (seq position)
  const int tid = threadIdx.x;

#pragma unroll
  for (int i = 0; i < 16; ++i) {
    int idx = i * 256 + tid;     // 32*128 = 4096 elems
    int t = idx >> 7, c = idx & 127;
    unsigned base = (unsigned)(g * 32 + t) * 384u;
    Ks[t][c] = qkv[base + 128 + c];
    Vs[t][c] = qkv[base + 256 + c];
  }
  __syncthreads();

  const int hd = tid >> 5;       // head = wave id
  const int l  = tid & 31;       // query index within group

  float q[16];
  unsigned qbase = (unsigned)(g * 32 + l) * 384u + hd * 16;
#pragma unroll
  for (int d = 0; d < 16; ++d) q[d] = qkv[qbase + d];

  float sc[32];
  float mx = -1e30f;
#pragma unroll
  for (int j = 0; j < 32; ++j) {
    float s = 0.0f;
#pragma unroll
    for (int d = 0; d < 16; ++d) s += q[d] * Ks[j][hd * 16 + d];
    s *= 0.25f;                  // 1/sqrt(DH), DH = 16
    sc[j] = s;
    mx = fmaxf(mx, s);
  }
  float sum = 0.0f;
#pragma unroll
  for (int j = 0; j < 32; ++j) { sc[j] = __expf(sc[j] - mx); sum += sc[j]; }
  float inv = 1.0f / sum;

  float o[16];
#pragma unroll
  for (int d = 0; d < 16; ++d) o[d] = 0.0f;
#pragma unroll
  for (int j = 0; j < 32; ++j) {
    float a = sc[j] * inv;
#pragma unroll
    for (int d = 0; d < 16; ++d) o[d] += a * Vs[j][hd * 16 + d];
  }
  unsigned obase = (unsigned)(g * 32 + l) * 128u + hd * 16;
#pragma unroll
  for (int d = 0; d < 16; ++d) out[obase + d] = o[d];
}

// ---------------- fused residual add + LayerNorm (D=128, wave per token) ---
__global__ __launch_bounds__(256)
void k_add_ln(const float* __restrict__ xa, const float* __restrict__ xb,
              const float* __restrict__ g, const float* __restrict__ b,
              float* __restrict__ out, int T) {
  int wave = threadIdx.x >> 5, lane = threadIdx.x & 31;
  int t = blockIdx.x * 8 + wave;
  if (t >= T) return;
  unsigned base = (unsigned)t * 128u;
  float v[4];
  float s = 0.0f;
#pragma unroll
  for (int i = 0; i < 4; ++i) {
    v[i] = xa[base + lane * 4 + i] + xb[base + lane * 4 + i];
    s += v[i];
  }
#pragma unroll
  for (int off = 16; off > 0; off >>= 1) s += __shfl_xor(s, off, 32);
  float mean = s * (1.0f / 128.0f);
  float vs = 0.0f;
#pragma unroll
  for (int i = 0; i < 4; ++i) { float d = v[i] - mean; vs += d * d; }
#pragma unroll
  for (int off = 16; off > 0; off >>= 1) vs += __shfl_xor(vs, off, 32);
  float inv = rsqrtf(vs * (1.0f / 128.0f) + 1e-5f);
#pragma unroll
  for (int i = 0; i < 4; ++i) {
    int c = lane * 4 + i;
    out[base + c] = (v[i] - mean) * inv * g[c] + b[c];
  }
}

// ---------------- heads (p,q,v,d) with q quaternion normalization ----------
__global__ __launch_bounds__(256)
void k_head(const float* __restrict__ hT, int row0,
            const float* __restrict__ Wp, const float* __restrict__ bp,
            const float* __restrict__ Wq, const float* __restrict__ bq,
            const float* __restrict__ Wv, const float* __restrict__ bv,
            const float* __restrict__ Wd, const float* __restrict__ bd,
            float* __restrict__ out) {
  __shared__ float res[32][26];
  int tid = threadIdx.x;
  for (int t = tid; t < 832; t += 256) {
    int b = t / 26, o = t % 26;
    const float* W; const float* bb; int row;
    if (o < 6)       { W = Wp; bb = bp; row = o; }
    else if (o < 14) { W = Wq; bb = bq; row = o - 6; }
    else if (o < 20) { W = Wv; bb = bv; row = o - 14; }
    else             { W = Wd; bb = bd; row = o - 20; }
    const float* h = hT + (unsigned)(row0 + b) * 128u;
    float s = bb[row];
    for (int k = 0; k < 128; ++k) s += h[k] * W[row * 128 + k];
    res[b][o] = s;
  }
  __syncthreads();
  for (int t = tid; t < 832; t += 256) {
    int b = t / 26, o = t % 26;
    float val = res[b][o];
    int dst;
    if (o < 6) {
      dst = b * 6 + o;
    } else if (o < 14) {
      int grp = (o - 6) >> 2;     // 0: q[0:4], 1: q[4:8]
      float nrm = 0.0f;
      for (int i = 0; i < 4; ++i) { float x = res[b][6 + grp * 4 + i]; nrm += x * x; }
      val *= rsqrtf(nrm);
      dst = 192 + b * 8 + (o - 6);
    } else if (o < 20) {
      dst = 192 + 256 + b * 6 + (o - 14);
    } else {
      dst = 192 + 256 + 192 + b * 6 + (o - 20);
    }
    out[dst] = val;
  }
}

// ---------------------------------------------------------------------------
extern "C" void kernel_launch(void* const* d_in, const int* in_sizes, int n_in,
                              void* d_out, int out_size, void* d_ws, size_t ws_size,
                              hipStream_t stream) {
  (void)n_in; (void)out_size; (void)ws_size;

  const int N   = in_sizes[0] / 128;   // 32768 nodes
  const int E   = in_sizes[2] / 2;     // 524288 edges per relation
  const int SEQ = N / 32;              // 1024 (batch = 32)

  const float* xL = (const float*)d_in[0];
  const float* xR = (const float*)d_in[1];
  const int* ei[4] = { (const int*)d_in[2],   // LL
                       (const int*)d_in[3],   // RR
                       (const int*)d_in[4],   // LR
                       (const int*)d_in[5] }; // RL
  // weight index helpers (dict order: gnn[layer][LL,RR,LR,RL]=(Wl,bl,Wr))
  auto GW = [&](int layer, int rel, int part) -> const float* {
    return (const float*)d_in[7 + layer * 12 + rel * 3 + part];
  };
  auto TW = [&](int layer, int j) -> const float* {
    return (const float*)d_in[43 + layer * 12 + j];
  };
  auto HW = [&](int j) -> const float* { return (const float*)d_in[91 + j]; };

  // ---------------- workspace layout (floats) ----------------
  float* ws   = (float*)d_ws;
  long long NN = N;
  float* agg  = ws;               ws = agg  + NN * 128;
  float* cnt  = ws;               ws = cnt  + NN;
  float* xbL0 = ws;               ws = xbL0 + NN * 64;
  float* xbL1 = ws;               ws = xbL1 + NN * 64;
  float* xbR0 = ws;               ws = xbR0 + NN * 64;
  float* xbR1 = ws;               ws = xbR1 + NN * 64;
  float* hT   = ws;               ws = hT   + NN * 128;
  float* hmid = ws;               ws = hmid + NN * 128;
  float* tmp1 = ws;               ws = tmp1 + NN * 128;
  float* attn = ws;               ws = attn + NN * 128;
  float* qkvb = ws;               ws = qkvb + NN * 384;
  __bf16* ffh = (__bf16*)ws;      /* bf16 FF intermediate: NN*2048*2 bytes */

  auto gemm = [&](const float* A, const float* B, int bT, const float* bias,
                  float* C, int M, int Nc, int K, int acc, int relu) {
    if (bT) {                      // transformer GEMMs: N % 128 == 0
      dim3 grid(M / BM, Nc / 128);
      k_gemm_bf16<1, 2, 0, 0><<<grid, 256, 0, stream>>>(A, B, K, bias, C, Nc, K, acc, relu);
    } else {                       // GNN GEMMs: N == 64
      dim3 grid(M / BM, Nc / 64);
      k_gemm_bf16<0, 1, 0, 0><<<grid, 256, 0, stream>>>(A, B, Nc, bias, C, Nc, K, acc, relu);
    }
  };
  auto zero = [&](float* p, long long n) {
    k_zero<<<(unsigned)((n + TB - 1) / TB), TB, 0, stream>>>(p, n);
  };
  auto sage_mean = [&](const float* xsrc, const int* e, int shift) {
    zero(agg, (long long)N << shift);
    zero(cnt, N);
    long long tot = (long long)E << shift;
    k_sage_agg<<<(unsigned)((tot + TB - 1) / TB), TB, 0, stream>>>(xsrc, e, agg, cnt, E, shift);
    long long tn = (long long)N << shift;
    k_mean<<<(unsigned)((tn + TB - 1) / TB), TB, 0, stream>>>(agg, cnt, shift, tn);
  };

  // ---------------- GNN: 3 hetero-SAGE layers ----------------
  // rel index: LL=0, RR=1, LR=2, RL=3 ; tuple part: Wl=0, bl=1, Wr=2
  for (int l = 0; l < 3; ++l) {
    const int din   = (l == 0) ? 128 : 64;
    const int shift = (l == 0) ? 7 : 6;
    const float* curL = (l == 0) ? xL : ((l == 1) ? xbL0 : xbL1);
    const float* curR = (l == 0) ? xR : ((l == 1) ? xbR0 : xbR1);
    float* outL = (l == 1) ? xbL1 : xbL0;
    float* outR = (l == 1) ? xbR1 : xbR0;

    // dst = L : relations LL (src L) and RL (src R)
    gemm(curL, GW(l, 0, 2), 0, GW(l, 0, 1), outL, N, 64, din, 0, 0); // xL @ Wr_LL + bl_LL
    sage_mean(curL, ei[0], shift);
    gemm(agg,  GW(l, 0, 0), 0, nullptr,     outL, N, 64, din, 1, 0); // + mean_LL @ Wl_LL
    gemm(curL, GW(l, 3, 2), 0, GW(l, 3, 1), outL, N, 64, din, 1, 0); // + xL @ Wr_RL + bl_RL
    sage_mean(curR, ei[3], shift);
    gemm(agg,  GW(l, 3, 0), 0, nullptr,     outL, N, 64, din, 1, 1); // + mean_RL @ Wl_RL, relu

    // dst = R : relations RR (src R) and LR (src L)
    gemm(curR, GW(l, 1, 2), 0, GW(l, 1, 1), outR, N, 64, din, 0, 0);
    sage_mean(curR, ei[1], shift);
    gemm(agg,  GW(l, 1, 0), 0, nullptr,     outR, N, 64, din, 1, 0);
    gemm(curR, GW(l, 2, 2), 0, GW(l, 2, 1), outR, N, 64, din, 1, 0);
    sage_mean(curL, ei[2], shift);
    gemm(agg,  GW(l, 2, 0), 0, nullptr,     outR, N, 64, din, 1, 1);
  }

  // concat + [b,s,D] -> [s,b,D] permute : hT is [SEQ*32, 128], token = s*32+b
  {
    long long tot = (long long)N * 128;
    k_concat_permute<<<(unsigned)((tot + TB - 1) / TB), TB, 0, stream>>>(xbL0, xbR0, hT, N, SEQ);
  }

  // ---------------- Transformer: 4 post-norm encoder layers ----------------
  for (int t = 0; t < 4; ++t) {
    const float* Wqkv = TW(t, 0); const float* bqkv = TW(t, 1);
    const float* Wo   = TW(t, 2); const float* bo   = TW(t, 3);
    const float* W1   = TW(t, 4); const float* b1   = TW(t, 5);
    const float* W2   = TW(t, 6); const float* b2   = TW(t, 7);
    const float* g1   = TW(t, 8); const float* be1  = TW(t, 9);
    const float* g2   = TW(t, 10); const float* be2 = TW(t, 11);

    gemm(hT, Wqkv, 1, bqkv, qkvb, N, 384, 128, 0, 0);        // qkv = h @ Wqkv^T + b
    k_attn<<<SEQ, 256, 0, stream>>>(qkvb, attn);             // per-group MHA
    gemm(attn, Wo, 1, bo, tmp1, N, 128, 128, 0, 0);          // attn @ Wo^T + bo
    k_add_ln<<<N / 8, 256, 0, stream>>>(hT, tmp1, g1, be1, hmid, N);

    // FF1: relu(h @ W1^T + b1) stored as bf16 (halves HBM round trip)
    {
      dim3 grid(N / BM, 2048 / 128);
      k_gemm_bf16<1, 2, 0, 1><<<grid, 256, 0, stream>>>(hmid, W1, 128, b1, ffh,
                                                        2048, 128, 0, 1);
    }
    // FF2: (bf16 ff) @ W2^T + b2 -> f32
    {
      dim3 grid(N / BM, 128 / 128);
      k_gemm_bf16<1, 2, 1, 0><<<grid, 256, 0, stream>>>(ffh, W2, 2048, b2, tmp1,
                                                        128, 2048, 0, 0);
    }
    k_add_ln<<<N / 8, 256, 0, stream>>>(hmid, tmp1, g2, be2, hT, N);
  }

  // ---------------- heads on h[-1] = seq position SEQ-1 ----------------
  k_head<<<1, 256, 0, stream>>>(hT, (SEQ - 1) * 32,
                                HW(0), HW(1), HW(2), HW(3),
                                HW(4), HW(5), HW(6), HW(7),
                                (float*)d_out);
}
